// FullyConnectedTensorProduct_36386962932436
// MI455X (gfx1250) — matrix operand order repaired and checked by reading, not compiled, not run on hardware
//
#include <hip/hip_runtime.h>

typedef __attribute__((ext_vector_type(16))) __bf16 v16bf;
typedef __attribute__((ext_vector_type(8)))  float  v8f;

union Frag { v16bf v; uint4 q[2]; };

#if defined(__has_builtin) && __has_builtin(__builtin_amdgcn_sched_barrier)
#define SCHED_FENCE() __builtin_amdgcn_sched_barrier(0)
#else
#define SCHED_FENCE()
#endif

__device__ __forceinline__ unsigned short f2bf(float f) {
  unsigned u = __float_as_uint(f);
  u += 0x7fffu + ((u >> 16) & 1u);          // round-to-nearest-even
  return (unsigned short)(u >> 16);
}

// ---------------------------------------------------------------------------
// Kernel 1: numerically build the real-basis Wigner-3j tables (exactly the
// reference's Racah formula + complex->real change of basis), scaled by the
// path normalization sqrt((2l3+1)/fan)/sqrt(2l3+1) = 1/sqrt(fan).
// ---------------------------------------------------------------------------
struct cplx { double re, im; };
__device__ __forceinline__ cplx cmul(cplx a, cplx b) {
  return cplx{a.re * b.re - a.im * b.im, a.re * b.im + a.im * b.re};
}

__device__ void qmat(int l, cplx q[5][5]) {
  for (int a = 0; a < 5; ++a)
    for (int b = 0; b < 5; ++b) q[a][b] = cplx{0.0, 0.0};
  const double is2 = 0.70710678118654752440;
  for (int m = -l; m < 0; ++m) {
    q[l + m][l - m].re = is2;     // q[l+m, l+|m|] = 1/sqrt(2)
    q[l + m][l + m].im = -is2;    // q[l+m, l-|m|] = -i/sqrt(2)
  }
  q[l][l].re = 1.0;
  for (int m = 1; m <= l; ++m) {
    double s = (m & 1) ? -1.0 : 1.0;
    q[l + m][l + m].re = s * is2;
    q[l + m][l - m].im = s * is2;
  }
  if (l == 1) {                    // multiply by (-i)^1
    for (int a = 0; a < 5; ++a)
      for (int b = 0; b < 5; ++b) {
        double re = q[a][b].re, im = q[a][b].im;
        q[a][b].re = im; q[a][b].im = -re;
      }
  } else if (l == 2) {             // multiply by (-i)^2 = -1
    for (int a = 0; a < 5; ++a)
      for (int b = 0; b < 5; ++b) { q[a][b].re = -q[a][b].re; q[a][b].im = -q[a][b].im; }
  }
}

__device__ double su2cg(int j1, int j2, int j3, int m1, int m2, int m3) {
  if (m1 + m2 != m3) return 0.0;
  const double F[8] = {1., 1., 2., 6., 24., 120., 720., 5040.};
  int vmin = max(max(-j1 + j2 + m3, -j1 + m1), 0);
  int vmax = min(min(j2 + j3 + m1, j3 - j1 + j2), j3 + m3);
  if (vmax < vmin) return 0.0;
  double C = sqrt((double)(2 * j3 + 1) * F[j3 + j1 - j2] * F[j3 - j1 + j2] * F[j1 + j2 - j3]
                  * F[j3 + m3] * F[j3 - m3]
                  / (F[j1 + j2 + j3 + 1] * F[j1 - m1] * F[j1 + m1] * F[j2 - m2] * F[j2 + m2]));
  double S = 0.0;
  for (int v = vmin; v <= vmax; ++v)
    S += (((v + j2 + m2) & 1) ? -1.0 : 1.0) * F[j2 + j3 + m1 - v] * F[j1 - m1 + v]
         / (F[v] * F[j3 - j1 + j2 - v] * F[j3 + m3 - v] * F[v + j1 - j2 - m3]);
  return C * S;
}

__global__ void precompute_cg(float* __restrict__ cgtab) {
  int e = blockIdx.x * blockDim.x + threadIdx.x;
  if (e >= 363) return;
  const int L1[11] = {0,0,0,1,1,1,1,2,2,2,2};
  const int L2[11] = {0,1,2,0,1,1,2,0,1,2,2};
  const int L3[11] = {0,1,2,1,0,2,1,2,1,0,2};
  const int CG[12] = {0,1,10,35,44,53,98,143,168,213,238,363};
  int p = 0;
  while (e >= CG[p + 1]) ++p;
  int r = e - CG[p];
  int l1 = L1[p], l2 = L2[p], l3 = L3[p];
  int d2 = 2 * l2 + 1, d3 = 2 * l3 + 1;
  int a = r / (d2 * d3); r -= a * (d2 * d3);
  int b = r / d3;
  int c = r - b * d3;
  cplx Q1[5][5], Q2[5][5], Q3[5][5];
  qmat(l1, Q1); qmat(l2, Q2); qmat(l3, Q3);
  double sre = 0.0;
  for (int i = 0; i < 2 * l1 + 1; ++i)
    for (int k = 0; k < 2 * l2 + 1; ++k) {
      int m1 = i - l1, m2 = k - l2, m3 = m1 + m2;
      if (m3 < -l3 || m3 > l3) continue;
      int n = m3 + l3;
      double C = su2cg(l1, l2, l3, m1, m2, m3);
      if (C == 0.0) continue;
      cplx t = cmul(Q1[i][a], Q2[k][b]);
      cplx q3c{Q3[n][c].re, -Q3[n][c].im};
      t = cmul(t, q3c);
      sre += t.re * C;
    }
  double fan = (l3 == 0) ? 384.0 : 512.0;   // 128 * (#paths into irrep l3)
  cgtab[e] = (float)(sre / sqrt(fan));
}

// ---------------------------------------------------------------------------
// Kernel 2: weights f32 [path][u][w] -> bf16 transposed [path][w][u] so that
// WMMA B fragments are contiguous 16-element (32 byte) loads per lane.
// ---------------------------------------------------------------------------
__global__ void convert_weights(const float* __restrict__ w, unsigned short* __restrict__ wt) {
  int idx = blockIdx.x * 256 + threadIdx.x;
  if (idx >= 180224) return;
  int p = idx >> 14;
  int r = idx & 16383;
  int n = r >> 7, u = r & 127;
  wt[idx] = f2bf(w[(p << 14) + u * 128 + n]);
}

// ---------------------------------------------------------------------------
// Main kernel helpers
// ---------------------------------------------------------------------------
// LDS element offsets of the three x1 segments (bf16, u-stride padded to 136)
#define SX_L0 0
#define SX_L1 2176        // 16 * 1 * 136
#define SX_L2 8704        // SX_L1 + 16 * 3 * 136

// Stage the full 16 x 1152 x1 tile as bf16 into [m][i][u(pad 136)] per segment.
__device__ __forceinline__ void stage_all(unsigned short* sx, const float* __restrict__ x1,
                                          int b0, int tid) {
  for (int idx = tid; idx < 16 * 1152; idx += 256) {
    int m = idx / 1152;
    int e = idx - m * 1152;                 // position within row (coalesced)
    float f = x1[(size_t)(b0 + m) * 1152 + e];
    int dst;
    if (e < 128) {
      dst = SX_L0 + m * 136 + e;                          // l=0: i=0, u=e
    } else if (e < 512) {
      int r = e - 128, u = r / 3, i = r - 3 * u;          // l=1
      dst = SX_L1 + (m * 3 + i) * 136 + u;
    } else {
      int r = e - 512, u = r / 5, i = r - 5 * u;          // l=2
      dst = SX_L2 + (m * 5 + i) * 136 + u;
    }
    sx[dst] = f2bf(f);
  }
}

// One path: y[b, i, :16 cols] = x1_seg @ W_p, then acc[k] += cgx[b,i,k] * y
template<int D1, int D3>
__device__ __forceinline__ void run_path(const unsigned short* sxseg, const float* scgx_p,
                                         const unsigned short* wtp, int lane, int n0,
                                         v8f (&acc)[D3]) {
  const int q16 = lane & 15;      // A: batch row m ; B: output col n
  const int h   = lane >> 4;
  Frag B[4];
  #pragma unroll
  for (int kc = 0; kc < 4; ++kc) {
    // B 32x16 bf16: lanes 0-15 hold K=0..15, lanes 16-31 hold K=16..31 (contig)
    const uint4* bp = (const uint4*)(wtp + (size_t)(n0 + q16) * 128 + kc * 32 + h * 16);
    B[kc].q[0] = bp[0];
    B[kc].q[1] = bp[1];
  }
  #pragma unroll
  for (int i = 0; i < D1; ++i) {
    Frag A[4];
    #pragma unroll
    for (int kc = 0; kc < 4; ++kc) {
      // A 16x32 bf16: lane half h: elems 0-15 -> K=h*16+e (contig)
      const uint4* ap = (const uint4*)(sxseg + ((q16 * D1 + i) * 136 + kc * 32 + h * 16));
      A[kc].q[0] = ap[0];
      A[kc].q[1] = ap[1];     // +8 elements = +16 bytes
    }
    v8f y = {0.f, 0.f, 0.f, 0.f, 0.f, 0.f, 0.f, 0.f};
    #pragma unroll
    for (int kc = 0; kc < 4; ++kc)
      y = __builtin_amdgcn_wmma_f32_16x16x32_bf16(false, A[kc].v, false, B[kc].v,
                                                  (short)0, y, false, false);
    // epilogue: C tile row m = h*8 + f
    #pragma unroll
    for (int f = 0; f < 8; ++f) {
      const float* cx = scgx_p + (h * 8 + f) * 115 + i * D3;
      #pragma unroll
      for (int k = 0; k < D3; ++k)
        acc[k][f] += cx[k] * y[f];
    }
  }
}

template<int D3>
__device__ __forceinline__ void write_out(float* __restrict__ out, int b0, int o3,
                                          int n0, int lane, const v8f (&acc)[D3]) {
  const int nn = n0 + (lane & 15);
  const int h  = lane >> 4;
  #pragma unroll
  for (int k = 0; k < D3; ++k)
    #pragma unroll
    for (int f = 0; f < 8; ++f)
      out[(size_t)(b0 + h * 8 + f) * 1152 + o3 + nn * D3 + k] = acc[k][f];
}

// ---------------------------------------------------------------------------
// Main kernel: 16 batch rows / block, 8 waves = 8 output-channel tiles of 16.
// All of x1 staged once; paths grouped by OUTPUT irrep so each group's C
// tiles (<=5) are written out and die before the next group starts.
// ---------------------------------------------------------------------------
__global__ __launch_bounds__(256) void fctp_wmma_kernel(
    const float* __restrict__ x1, const float* __restrict__ x2,
    const float* __restrict__ cgtab, const unsigned short* __restrict__ wt,
    float* __restrict__ out) {
  __shared__ __align__(16) unsigned short sx[16 * 9 * 136];   // 39168 B, all 3 segments
  __shared__ float scgx[16 * 115];                            // 7360 B

  const int tid  = threadIdx.x;
  const int b0   = blockIdx.x * 16;
  const int lane = tid & 31;
  const int n0   = (tid >> 5) * 16;

  stage_all(sx, x1, b0, tid);

  // cgx[b, i, k] = sum_j cg[i,j,k] * x2[b, j]  (tiny, 176 threads)
  if (tid < 176) {
    const int D1t[11] = {1,1,1,3,3,3,3,5,5,5,5};
    const int D2t[11] = {1,3,5,1,3,3,5,1,3,5,5};
    const int D3t[11] = {1,3,5,3,1,5,3,5,3,1,5};
    const int O2t[11] = {0,1,4,0,1,1,4,0,1,4,4};
    const int CGt[11] = {0,1,10,35,44,53,98,143,168,213,238};
    const int CXt[11] = {0,1,4,9,18,21,36,45,70,85,90};
    int m = tid / 11;
    int p = tid - m * 11;
    int d1 = D1t[p], d2 = D2t[p], d3 = D3t[p];
    const float* cg = cgtab + CGt[p];
    const float* xr = x2 + (size_t)(b0 + m) * 9 + O2t[p];
    float xs[5];
    for (int j = 0; j < d2; ++j) xs[j] = xr[j];
    float* dst = scgx + m * 115 + CXt[p];
    for (int i = 0; i < d1; ++i)
      for (int k = 0; k < d3; ++k) {
        float s = 0.f;
        for (int j = 0; j < d2; ++j) s += cg[(i * d2 + j) * d3 + k] * xs[j];
        dst[i * d3 + k] = s;
      }
  }
  __syncthreads();

  const v8f vzero = {0.f, 0.f, 0.f, 0.f, 0.f, 0.f, 0.f, 0.f};

  // ---- output irrep l3 = 0 (offset 0): paths (0,0,0), (1,1,0), (2,2,0) ----
  {
    v8f acc[1];
    acc[0] = vzero;
    run_path<1, 1>(sx + SX_L0, scgx + 0,  wt + 0 * 16384, lane, n0, acc);
    run_path<3, 1>(sx + SX_L1, scgx + 18, wt + 4 * 16384, lane, n0, acc);
    run_path<5, 1>(sx + SX_L2, scgx + 85, wt + 9 * 16384, lane, n0, acc);
    write_out<1>(out, b0, 0, n0, lane, acc);
  }
  SCHED_FENCE();

  // ---- output irrep l3 = 1 (offset 128): (0,1,1), (1,0,1), (1,2,1), (2,1,1) ----
  {
    v8f acc[3];
    #pragma unroll
    for (int k = 0; k < 3; ++k) acc[k] = vzero;
    run_path<1, 3>(sx + SX_L0, scgx + 1,  wt + 1 * 16384, lane, n0, acc);
    run_path<3, 3>(sx + SX_L1, scgx + 9,  wt + 3 * 16384, lane, n0, acc);
    run_path<3, 3>(sx + SX_L1, scgx + 36, wt + 6 * 16384, lane, n0, acc);
    run_path<5, 3>(sx + SX_L2, scgx + 70, wt + 8 * 16384, lane, n0, acc);
    write_out<3>(out, b0, 128, n0, lane, acc);
  }
  SCHED_FENCE();

  // ---- output irrep l3 = 2 (offset 512): (0,2,2), (1,1,2), (2,0,2), (2,2,2) ----
  {
    v8f acc[5];
    #pragma unroll
    for (int k = 0; k < 5; ++k) acc[k] = vzero;
    run_path<1, 5>(sx + SX_L0, scgx + 4,  wt + 2 * 16384,  lane, n0, acc);
    run_path<3, 5>(sx + SX_L1, scgx + 21, wt + 5 * 16384,  lane, n0, acc);
    run_path<5, 5>(sx + SX_L2, scgx + 45, wt + 7 * 16384,  lane, n0, acc);
    run_path<5, 5>(sx + SX_L2, scgx + 90, wt + 10 * 16384, lane, n0, acc);
    write_out<5>(out, b0, 512, n0, lane, acc);
  }
}

extern "C" void kernel_launch(void* const* d_in, const int* in_sizes, int n_in,
                              void* d_out, int out_size, void* d_ws, size_t ws_size,
                              hipStream_t stream) {
  (void)in_sizes; (void)n_in; (void)out_size; (void)ws_size;
  const float* x1 = (const float*)d_in[0];
  const float* x2 = (const float*)d_in[1];
  const float* w  = (const float*)d_in[2];
  float* out = (float*)d_out;

  float* cgtab = (float*)d_ws;                                   // 363 f32
  unsigned short* wt = (unsigned short*)((char*)d_ws + 4096);    // 180224 bf16

  precompute_cg<<<1, 384, 0, stream>>>(cgtab);
  convert_weights<<<704, 256, 0, stream>>>(w, wt);
  fctp_wmma_kernel<<<4096, 256, 0, stream>>>(x1, x2, cgtab, wt, out);
}
